// SkeletonResidualTranspose_9225589752530
// MI455X (gfx1250) — matrix operand order repaired and checked by reading, not compiled, not run on hardware
//
#include <hip/hip_runtime.h>

// ---- types -----------------------------------------------------------------
typedef __attribute__((ext_vector_type(16))) __bf16 v16bf;
typedef __attribute__((ext_vector_type(8)))  float  v8f;
typedef __attribute__((ext_vector_type(4)))  unsigned int u32x4; // trivial uint4

#define JOINTS   25
#define T_IN     1024
#define T_OUT    2048
#define C_UP     416      // 13*32 pooled input channels
#define CPJ_IN   32
#define CPJ_OUT  16
#define OUT_CH   400
#define IN_CH    800
#define BATCH    16
#define NSTEPS   21       // 7 taps * 3 channel-chunks of 32  (K = 672)
#define TSLAB    128      // time steps per workgroup (4 waves x 32)
#define LDS_COLS 144      // time cols staged: t in [tgg-8, tgg+135]
#define LDS_PITCH 72      // halves per col (64 ch + 8 pad) -> conflict-free b128 reads

static __device__ __forceinline__ unsigned short f2bf(float f) {
  unsigned int u = __float_as_uint(f);
  u += 0x7FFFu + ((u >> 16) & 1u);          // round-to-nearest-even
  return (unsigned short)(u >> 16);
}

// ---- kernel 1: linear 2x upsample, f32 -> bf16 -----------------------------
__global__ __launch_bounds__(256) void k_upsample(const float* __restrict__ x,
                                                  unsigned short* __restrict__ up) {
  const int row = blockIdx.x;                       // b*416 + c
  const float* xr = x + (size_t)row * T_IN;
  unsigned short* ur = up + (size_t)row * T_OUT;
  const int k0 = threadIdx.x * 4;                   // outputs 8*tid .. 8*tid+7
  float xv[6];
#pragma unroll
  for (int i = 0; i < 6; ++i) {
    int idx = k0 - 1 + i;
    idx = idx < 0 ? 0 : (idx > T_IN - 1 ? T_IN - 1 : idx);
    xv[i] = xr[idx];                                // clamp reproduces boundary exactly
  }
  union { u32x4 q; unsigned short h[8]; } o;
#pragma unroll
  for (int jj = 0; jj < 4; ++jj) {
    float xm = xv[jj], xc = xv[jj + 1], xp = xv[jj + 2];
    o.h[2 * jj]     = f2bf(0.25f * xm + 0.75f * xc);
    o.h[2 * jj + 1] = f2bf(0.75f * xc + 0.25f * xp);
  }
  *(u32x4*)(ur + (size_t)threadIdx.x * 8) = o.q;
}

// ---- kernel 2: pack masked weights into per-lane WMMA A-fragment layout ----
// A 16x32 bf16 layout (ISA 7.12.2): lane m (0..15): K = {0..7,16..23};
// lane m+16: K = {8..15,24..31}.  Shortcut folded into tap 3.
__global__ __launch_bounds__(32) void k_pack_w(const float* __restrict__ w_res,
                                               const float* __restrict__ w_sc,
                                               unsigned short* __restrict__ wpack) {
  const int js = blockIdx.x;                        // j*21 + s
  const int j = js / NSTEPS, s = js % NSTEPS;
  const int tap = s / 3, chunk = s % 3;
  int base = j - 1; base = base < 0 ? 0 : base; base = base > 22 ? 22 : base;
  const int block = base + chunk;
  const bool allowed = (block >= j - 1) && (block <= j + 1);  // block mask
  const int lane = threadIdx.x;
  const int m = lane & 15, hi = lane >> 4;
  const int oc = j * CPJ_OUT + m;
  union { u32x4 q[2]; unsigned short h[16]; } a;
#pragma unroll
  for (int h = 0; h < 16; ++h) {
    const int Kl = h + hi * 8 + (h >= 8 ? 8 : 0);  // channel index within 32-chunk
    const int ic = block * CPJ_IN + Kl;
    float v = 0.f;
    if (allowed) {
      v = w_res[((size_t)oc * IN_CH + ic) * 7 + tap];
      if (tap == 3) v += w_sc[(size_t)oc * IN_CH + ic];
    }
    a.h[h] = f2bf(v);
  }
  u32x4* dst = (u32x4*)wpack + ((size_t)js * 32 + lane) * 2;
  dst[0] = a.q[0];
  dst[1] = a.q[1];
}

// ---- kernel 2b: pre-sum biases ---------------------------------------------
__global__ __launch_bounds__(256) void k_bias(const float* __restrict__ b_res,
                                              const float* __restrict__ b_sc,
                                              float* __restrict__ bias) {
  const int i = blockIdx.x * 256 + threadIdx.x;
  if (i < OUT_CH) bias[i] = b_res[i] + b_sc[i];
}

// ---- kernel 3: block-sparse conv GEMM via v_wmma_f32_16x16x32_bf16 ---------
// WG = (128-time slab, joint, batch); 4 waves, each computes TWO 16x16 D tiles
// (M = 16 oc of joint, N = 2 x 16 time steps) sharing one A fragment per step:
// per WMMA cost = 1 global b128 (A) + 2 ds b128 (B).
__global__ __launch_bounds__(128) void k_gemm(const unsigned short* __restrict__ up,
                                              const unsigned short* __restrict__ wpack,
                                              const float* __restrict__ bias,
                                              const float* __restrict__ prelu,
                                              float* __restrict__ out) {
  __shared__ unsigned short lds[LDS_COLS * LDS_PITCH];
  const int tgrp = blockIdx.x;                      // 0..15
  const int j    = blockIdx.y;                      // 0..24
  const int b    = blockIdx.z;                      // 0..15
  const int tid  = threadIdx.x;
  const int lane = tid & 31, wv = tid >> 5;
  const int tgg  = tgrp * TSLAB;
  int base = j - 1; base = base < 0 ? 0 : base; base = base > 22 ? 22 : base;
  const int u0 = base >> 1;                         // first of 2 up-blocks (unpool gather)
  const unsigned short* uprow =
      up + ((size_t)b * C_UP + (size_t)u0 * CPJ_IN) * T_OUT;

  // Stage 64 channels x 144 cols (t in [tgg-8, tgg+135]) transposed into LDS.
#pragma unroll
  for (int it = 0; it < 9; ++it) {
    const int i  = tid + it * 128;                  // 0..1151
    const int g  = i % 18, ch = i / 18;
    const int t0 = tgg - 8 + g * 8;                 // 16B-aligned when interior
    union { u32x4 q; unsigned short h[8]; } v;
    if (t0 >= 0 && t0 + 7 < T_OUT) {
      v.q = *(const u32x4*)(uprow + (size_t)ch * T_OUT + t0);
    } else {
#pragma unroll
      for (int e = 0; e < 8; ++e) {
        int ts = t0 + e;
        ts = ts < 0 ? -ts : ts;                       // reflect left
        ts = ts >= T_OUT ? (2 * T_OUT - 2 - ts) : ts; // reflect right
        v.h[e] = uprow[(size_t)ch * T_OUT + ts];
      }
    }
#pragma unroll
    for (int e = 0; e < 8; ++e)
      lds[(g * 8 + e) * LDS_PITCH + ch] = v.h[e];
  }
  __syncthreads();

  const int n  = lane & 15, hi = lane >> 4;
  const int kb = hi << 3;                           // B frag: lanes 16-31 hold K={8..15,24..31}
  const int off1 = (base & 1) << 5;                 // LDS ch-offset of middle block
  v8f acc0 = {0.f, 0.f, 0.f, 0.f, 0.f, 0.f, 0.f, 0.f};
  v8f acc1 = {0.f, 0.f, 0.f, 0.f, 0.f, 0.f, 0.f, 0.f};
  const u32x4* wp = (const u32x4*)wpack + ((size_t)j * NSTEPS * 32 + lane) * 2;

#pragma unroll
  for (int s = 0; s < NSTEPS; ++s) {
    const int tap = s / 3, chunk = s % 3;
    const int choff = (chunk == 0) ? 0 : (chunk == 1 ? off1 : 32);
    const int col = (wv << 5) + n + tap + 5;        // t + tap - 3 relative to tgg-8
    union { u32x4 q[2]; v16bf v; } Af, B0, B1;
    Af.q[0] = wp[s * 64];
    Af.q[1] = wp[s * 64 + 1];
    const unsigned short* p0 = &lds[col * LDS_PITCH + choff + kb];
    B0.q[0] = *(const u32x4*)p0;                    // 16B aligned, conflict-free
    B0.q[1] = *(const u32x4*)(p0 + 16);
    const unsigned short* p1 = p0 + 16 * LDS_PITCH; // second N tile: t + 16
    B1.q[0] = *(const u32x4*)p1;
    B1.q[1] = *(const u32x4*)(p1 + 16);
    acc0 = __builtin_amdgcn_wmma_f32_16x16x32_bf16(false, Af.v, false, B0.v,
                                                   (short)0, acc0, false, false);
    acc1 = __builtin_amdgcn_wmma_f32_16x16x32_bf16(false, Af.v, false, B1.v,
                                                   (short)0, acc1, false, false);
  }

  // Epilogue: fused bias + PReLU, coalesced stores (16 consecutive t per row).
  const float alpha = prelu[0];
  const int t0 = tgg + (wv << 5) + n;
  float* orow = out + ((size_t)b * OUT_CH + (size_t)j * CPJ_OUT) * T_OUT + t0;
#pragma unroll
  for (int v = 0; v < 8; ++v) {
    const int oc = v + (hi << 3);                   // D layout: M = v + 8*hi
    const float bs = bias[j * CPJ_OUT + oc];
    float v0 = acc0[v] + bs;
    float v1 = acc1[v] + bs;
    v0 = v0 >= 0.f ? v0 : alpha * v0;
    v1 = v1 >= 0.f ? v1 : alpha * v1;
    orow[(size_t)oc * T_OUT]      = v0;
    orow[(size_t)oc * T_OUT + 16] = v1;
  }
}

// ---- host launcher ---------------------------------------------------------
extern "C" void kernel_launch(void* const* d_in, const int* in_sizes, int n_in,
                              void* d_out, int out_size, void* d_ws, size_t ws_size,
                              hipStream_t stream) {
  const float* x     = (const float*)d_in[0];  // (16, 416, 1024)
  const float* w_res = (const float*)d_in[1];  // (400, 800, 7)
  const float* b_res = (const float*)d_in[2];  // (400,)
  const float* w_sc  = (const float*)d_in[3];  // (400, 800, 1)
  const float* b_sc  = (const float*)d_in[4];  // (400,)
  const float* prelu = (const float*)d_in[5];  // (1,)
  float* out = (float*)d_out;                  // (16, 400, 2048)

  unsigned short* up    = (unsigned short*)d_ws;             // 16*416*2048 bf16 = 27.3 MB
  unsigned short* wpack = up + (size_t)BATCH * C_UP * T_OUT; // 25*21*512 bf16 = 0.54 MB
  float* bias = (float*)(wpack + (size_t)JOINTS * NSTEPS * 512); // 400 floats

  k_upsample<<<dim3(BATCH * C_UP), dim3(256), 0, stream>>>(x, up);
  k_pack_w  <<<dim3(JOINTS * NSTEPS), dim3(32), 0, stream>>>(w_res, w_sc, wpack);
  k_bias    <<<dim3(2), dim3(256), 0, stream>>>(b_res, b_sc, bias);
  k_gemm    <<<dim3(T_OUT / TSLAB, JOINTS, BATCH), dim3(128), 0, stream>>>(
      up, wpack, bias, prelu, out);
}
